// LSTM_22832046145965
// MI455X (gfx1250) — matrix operand order repaired and checked by reading, not compiled, not run on hardware
//
#include <hip/hip_runtime.h>

// LSTM  B=32 T=1024 D=512 H=512  (4H = 2048)
// Phase 1: Zx = X @ Wx, t-major, TDM-staged LDS tiles, f16 WMMA, output in
//          WMMA-fragment-packed layout (one b128 store per lane).
// Phase 2: persistent 16-WG recurrent kernel. Wh slice (128KB f16) resident
//          in CDNA5 320KB LDS. h exchanged via global f16 buffer, gathered
//          per step with one TDM load (hw row padding). Cluster barrier +
//          deterministic atomic grid barrier per step.
// Phase 3: gather final states by lengths (int64).

typedef __attribute__((ext_vector_type(16))) _Float16 v16h;
typedef __attribute__((ext_vector_type(8)))  float    v8f;
typedef __attribute__((ext_vector_type(4)))  unsigned u32x4_t;
typedef __attribute__((ext_vector_type(8)))  int      i32x8_t;
typedef __attribute__((ext_vector_type(4)))  int      i32x4_t;
typedef _Float16 half_t;

#define Bb 32
#define Tt 1024
#define Dd 512
#define Hh 512
#define ZC 2048
#define NWG 16

#if defined(__HIP_DEVICE_COMPILE__) && __has_builtin(__builtin_amdgcn_tensor_load_to_lds) && \
    __has_builtin(__builtin_amdgcn_s_wait_tensorcnt)
#define HAVE_TDM 1
#else
#define HAVE_TDM 0
#endif

union Frag16 { v16h v; _Float16 h[16]; unsigned u[8]; };
union Pack8  { uint4 q; _Float16 h[8]; };

// Fragment-packed Zx layout: idx(t, mt, zcol, hi, r) -> 8 contiguous f16.
__device__ __forceinline__ size_t zx_idx(int t, int mt, int zcol, int hi) {
  return ((((size_t)t * 2 + mt) * ZC + zcol) * 2 + hi) * 8;
}

#if HAVE_TDM
// Build D# (groups 0..3) for an up-to-3D tile load, issue TENSOR_LOAD_TO_LDS.
// Address = global + dsz*(x + y*s0 + z*s1). Optional LDS row padding.
__device__ __forceinline__ void tdm_load(
    unsigned lds_off, const void* gptr, unsigned dsz_code,
    unsigned tile0, unsigned tile1, unsigned tile2,
    unsigned dim0, unsigned dim1, unsigned dim2,
    unsigned long long s0, unsigned long long s1,
    unsigned pad_int, unsigned pad_amt, bool pad_en) {
  unsigned long long ga = (unsigned long long)(uintptr_t)gptr;
  u32x4_t g0;
  g0.x = 1u;                                             // count=1, user mode
  g0.y = lds_off;                                        // lds_addr (bytes)
  g0.z = (unsigned)(ga & 0xFFFFFFFFu);                   // global_addr[31:0]
  g0.w = (unsigned)((ga >> 32) & 0x01FFFFFFu) | 0x80000000u;  // [56:32]|type=2
  unsigned w0 = (dsz_code << 16);
  if (pad_en) w0 |= (1u << 20) | (pad_int << 22) | (pad_amt << 25);
  i32x8_t g1;
  g1[0] = (int)w0;                                       // mask=0,dsz,pad
  g1[1] = (int)((dim0 & 0xFFFFu) << 16);                 // dim0[15:0]
  g1[2] = (int)((dim0 >> 16) | ((dim1 & 0xFFFFu) << 16));
  g1[3] = (int)((dim1 >> 16) | (tile0 << 16));
  g1[4] = (int)((tile1 & 0xFFFFu) | (tile2 << 16));
  g1[5] = (int)(unsigned)(s0 & 0xFFFFFFFFull);           // dim0_stride lo
  g1[6] = (int)((unsigned)((s0 >> 32) & 0xFFFFull) | (unsigned)((s1 & 0xFFFFull) << 16));
  g1[7] = (int)(unsigned)((s1 >> 16) & 0xFFFFFFFFull);   // dim1_stride hi
  i32x4_t g2 = {(int)dim2, 0, 0, 0};
  i32x4_t g3 = {0, 0, 0, 0};
#if __clang_major__ >= 23
  i32x8_t g4 = {0, 0, 0, 0, 0, 0, 0, 0};
  __builtin_amdgcn_tensor_load_to_lds(g0, g1, g2, g3, g4, 0);
#else
  __builtin_amdgcn_tensor_load_to_lds(g0, g1, g2, g3, 0);
#endif
}
#endif

// ---------------------------------------------------------------- phase 1 ---
// Block: 8 waves, covers 2 timesteps x 32 batches x 64 z-cols.
// Wave wv: mt = wv&1 (batch half), ctl = wv>>1; each wave owns two 16x16
// tiles (dt = 0,1) sharing one B fragment per k-step.
// LDS (dynamic): 2x A buf [64][66] f32, 2x Wx buf [64][66] f32 (pad 2 dw/row).
__global__ __launch_bounds__(256) void lstm_xproj(const float* __restrict__ X,
                                                  const float* __restrict__ Wx,
                                                  half_t* __restrict__ Zxp) {
  extern __shared__ char smem1[];
  float* const sA0 = (float*)smem1;
  float* const sA1 = (float*)(smem1 + 16896);
  float* const sW0 = (float*)(smem1 + 2 * 16896);
  float* const sW1 = (float*)(smem1 + 3 * 16896);

  const int tid = threadIdx.x;
  const int wv = tid >> 5, lane = tid & 31;
  const int lo = lane & 15, hi = lane >> 4;
  const int mt = wv & 1;            // batch half (tile rows = 16 batches)
  const int ctl = wv >> 1;          // 0..3 -> 16-col tile within 64-col slab
  const int cg = blockIdx.x & 31;   // 32 col groups of 64
  const int t0 = (blockIdx.x >> 5) * 2;
  const int c0 = cg * 64;

#if HAVE_TDM
  if (wv == 0) {
    // chunk 0: X tile 64k x 2t x 32b, strides 512 (t), T*D (b); pad 64+2 dw
    tdm_load((unsigned)(uintptr_t)sA0, X + (size_t)t0 * Dd, /*dsz 4B*/2,
             64, 2, 32, 64, 2, 32, (unsigned long long)Dd,
             (unsigned long long)Tt * Dd, 5, 1, true);
    // chunk 0: Wx tile 64c x 64k, stride 2048
    tdm_load((unsigned)(uintptr_t)sW0, Wx + c0, 2,
             64, 64, 0, 64, 64, 0, (unsigned long long)ZC, 0ull, 5, 1, true);
  }
#endif

  v8f acc0 = {}, acc1 = {};         // dt = 0, 1
  for (int ck = 0; ck < 8; ++ck) {
    const int p = ck & 1;
    const int kc0 = ck * 64;
    float* const sAp = p ? sA1 : sA0;
    float* const sWp = p ? sW1 : sW0;
    float* const sAn = p ? sA0 : sA1;   // next buffer
    float* const sWn = p ? sW0 : sW1;
    __syncthreads();                // next buffer free before TDM refill
#if HAVE_TDM
    if (wv == 0) {
      if (ck + 1 < 8) {
        const int kn = kc0 + 64;
        tdm_load((unsigned)(uintptr_t)sAn, X + (size_t)t0 * Dd + kn, 2,
                 64, 2, 32, 64, 2, 32, (unsigned long long)Dd,
                 (unsigned long long)Tt * Dd, 5, 1, true);
        tdm_load((unsigned)(uintptr_t)sWn, Wx + (size_t)kn * ZC + c0, 2,
                 64, 64, 0, 64, 64, 0, (unsigned long long)ZC, 0ull, 5, 1, true);
        __builtin_amdgcn_s_wait_tensorcnt(2);   // chunk ck landed
      } else {
        __builtin_amdgcn_s_wait_tensorcnt(0);
      }
    }
#else
    // Fallback staging. LDS row order for A: row = b*2 + dt.
    for (int i = tid; i < 64 * 64; i += 256) {
      const int r = i >> 6, kk = i & 63;
      sAp[r * 66 + kk] =
          X[((size_t)(r >> 1) * Tt + t0 + (r & 1)) * Dd + kc0 + kk];
      sWp[r * 66 + kk] = Wx[(size_t)(kc0 + r) * ZC + c0 + kk];
    }
#endif
    __syncthreads();
#pragma unroll
    for (int ks = 0; ks < 2; ++ks) {
      const int kb0 = ks * 32;
      Frag16 a0, a1, bf;
#pragma unroll
      for (int v = 0; v < 8; ++v) {
        const int kb = kb0 + ((v < 4) ? 0 : 16) + hi * 8 + (v & 3) * 2;
        const int rowb = (mt * 16 + lo) * 2;          // b*2 + dt
        const float2 fa0 = *(const float2*)(sAp + (size_t)(rowb + 0) * 66 + kb);
        const float2 fa1 = *(const float2*)(sAp + (size_t)(rowb + 1) * 66 + kb);
        a0.h[2 * v] = (_Float16)fa0.x; a0.h[2 * v + 1] = (_Float16)fa0.y;
        a1.h[2 * v] = (_Float16)fa1.x; a1.h[2 * v + 1] = (_Float16)fa1.y;
        const int kk = kb0 + ((v < 4) ? 0 : 16) + hi * 8 + (v & 3) * 2;
        const int cl = ctl * 16 + lo;
        bf.h[2 * v]     = (_Float16)sWp[(size_t)kk * 66 + cl];
        bf.h[2 * v + 1] = (_Float16)sWp[(size_t)(kk + 1) * 66 + cl];
      }
      acc0 = __builtin_amdgcn_wmma_f32_16x16x32_f16(false, a0.v, false, bf.v,
                                                    (short)0, acc0, false, false);
      acc1 = __builtin_amdgcn_wmma_f32_16x16x32_f16(false, a1.v, false, bf.v,
                                                    (short)0, acc1, false, false);
    }
  }
  // Fragment-packed store: one b128 per lane per dt.
  const int zc = c0 + ctl * 16 + lo;
  Pack8 o0, o1;
#pragma unroll
  for (int r = 0; r < 8; ++r) { o0.h[r] = (_Float16)acc0[r]; o1.h[r] = (_Float16)acc1[r]; }
  *(uint4*)(Zxp + zx_idx(t0 + 0, mt, zc, hi)) = o0.q;
  *(uint4*)(Zxp + zx_idx(t0 + 1, mt, zc, hi)) = o1.q;
}

// ------------------------------------------------------- grid barrier ------
__device__ __forceinline__ void grid_barrier(unsigned* ctr, unsigned* target) {
  __syncthreads();
  *target += NWG;
  if (threadIdx.x == 0) {
    __hip_atomic_fetch_add(ctr, 1u, __ATOMIC_ACQ_REL, __HIP_MEMORY_SCOPE_AGENT);
    while (__hip_atomic_load(ctr, __ATOMIC_ACQUIRE,
                             __HIP_MEMORY_SCOPE_AGENT) < *target)
      __builtin_amdgcn_s_sleep(1);
  }
  // CDNA5 cluster barrier (s_barrier_signal/wait -3); NOP off-cluster, in
  // which case the atomic barrier above carries the sync.
  __builtin_amdgcn_s_cluster_barrier();
  __syncthreads();
}

// ---------------------------------------------------------------- phase 2 ---
__global__ __launch_bounds__(256) void lstm_rec(
    const half_t* __restrict__ Zxp, const float* __restrict__ Wh,
    const float* __restrict__ bias, float* __restrict__ out,
    half_t* __restrict__ hbuf, unsigned* __restrict__ ctr) {
  extern __shared__ char smem[];
  half_t* sWh = (half_t*)smem;                              // [128][520] f16
  half_t* sh  = (half_t*)(smem + 128 * 520 * 2);            // [32][520]  f16
  float* zbuf = (float*)(smem + 128 * 520 * 2 + 32 * 520 * 2); // [32][132]
  float* cst  = zbuf + 32 * 132;                            // [32][32]

  const int wg = blockIdx.x;
  const int tid = threadIdx.x;
  const int wv = tid >> 5, lane = tid & 31;
  const int lo = lane & 15, hi = lane >> 4;
  const int col0 = wg * 32;

  for (int i = tid; i < 128 * 512; i += 256) {   // Wh slice, LDS-resident
    const int c = i >> 9, k = i & 511;
    const int g = c >> 5, j = c & 31;
    sWh[(size_t)c * 520 + k] = (half_t)Wh[(size_t)k * ZC + g * Hh + col0 + j];
  }
  for (int i = tid; i < 32 * 32; i += 256) cst[i] = 0.f;
  for (int i = tid; i < 32 * 32; i += 256) {
    const int m = i >> 5, j = i & 31;
    hbuf[m * Hh + col0 + j] = (half_t)0.f;       // h0 = 0
  }
  __threadfence();
  unsigned target = 0;
  grid_barrier(ctr, &target);

  const int lc = wv * 16 + lo;                   // local z col 0..127
  const int zcol = (lc >> 5) * Hh + col0 + (lc & 31);
  const float bia = bias[zcol];

  for (int t = 0; t < Tt; ++t) {
    // ---- gather h(t-1): one TDM load, hw-padded rows (512 -> 520 halfs)
#if HAVE_TDM
    if (wv == 0) {
      tdm_load((unsigned)(uintptr_t)sh, hbuf, /*dsz 2B*/1,
               512, 32, 0, 512, 32, 0, 512ull, 0ull, /*256dw*/7, /*4dw*/3, true);
      __builtin_amdgcn_s_wait_tensorcnt(0);
    }
#else
    for (int i = tid; i < 32 * 128; i += 256) {
      const int m = i >> 7, q = i & 127;
      *(uint2*)(sh + (size_t)m * 520 + q * 4) =
          *(const uint2*)(hbuf + (size_t)m * Hh + q * 4);
    }
#endif
    grid_barrier(ctr, &target);   // all WGs consumed h(t-1)

    // ---- acc seeded with Zx + b: one b128 per M-tile per lane
    Pack8 p0, p1;
    p0.q = *(const uint4*)(Zxp + zx_idx(t, 0, zcol, hi));
    p1.q = *(const uint4*)(Zxp + zx_idx(t, 1, zcol, hi));
    v8f acc0, acc1;
#pragma unroll
    for (int r = 0; r < 8; ++r) {
      acc0[r] = (float)p0.h[r] + bia;
      acc1[r] = (float)p1.h[r] + bia;
    }
    if (t + 1 < Tt)
      __builtin_prefetch(Zxp + zx_idx(t + 1, 0, zcol, hi), 0, 1);

    // ---- z += h @ Wh (K=512; two M-tiles share the B fragment)
    for (int k0 = 0; k0 < Dd; k0 += 32) {
      Frag16 a0, a1, bf;
#pragma unroll
      for (int v = 0; v < 8; ++v) {
        const int kb = ((v < 4) ? 0 : 16) + hi * 8 + (v & 3) * 2;
        a0.u[v] = *(const unsigned*)(sh + (size_t)lo * 520 + k0 + kb);
        a1.u[v] = *(const unsigned*)(sh + (size_t)(16 + lo) * 520 + k0 + kb);
        const int kk = k0 + ((v < 4) ? 0 : 16) + hi * 8 + (v & 3) * 2;
        bf.u[v] = *(const unsigned*)(sWh + (size_t)lc * 520 + kk);
      }
      acc0 = __builtin_amdgcn_wmma_f32_16x16x32_f16(false, a0.v, false, bf.v,
                                                    (short)0, acc0, false, false);
      acc1 = __builtin_amdgcn_wmma_f32_16x16x32_f16(false, a1.v, false, bf.v,
                                                    (short)0, acc1, false, false);
    }
#pragma unroll
    for (int r = 0; r < 8; ++r) {
      zbuf[(hi * 8 + r) * 132 + lc]      = acc0[r];
      zbuf[(16 + hi * 8 + r) * 132 + lc] = acc1[r];
    }
    __syncthreads();

    // ---- fused gates + state update (c resident in LDS)
    for (int i = tid; i < 32 * 32; i += 256) {
      const int m = i >> 5, j = i & 31;
      const float zi = zbuf[m * 132 + j];
      const float zf = zbuf[m * 132 + 32 + j];
      const float zg = zbuf[m * 132 + 64 + j];
      const float zo = zbuf[m * 132 + 96 + j];
      const float ig = 1.f / (1.f + __expf(-zi));
      const float fg = 1.f / (1.f + __expf(-zf));
      const float gg = tanhf(zg);
      const float og = 1.f / (1.f + __expf(-zo));
      const float c = fg * cst[m * 32 + j] + ig * gg;
      cst[m * 32 + j] = c;
      const float h = og * tanhf(c);
      out[((size_t)m * Tt + t) * Hh + col0 + j] = h;
      hbuf[(size_t)m * Hh + col0 + j] = (half_t)h;
    }
    __threadfence();
    grid_barrier(ctr, &target);   // publish h(t)
  }
}

// ---------------------------------------------------------------- phase 3 ---
__global__ void lstm_init_ctr(unsigned* ctr) {
  if (threadIdx.x == 0) *ctr = 0u;
}

__global__ void lstm_final(const float* __restrict__ out,
                           const long long* __restrict__ lengths,
                           float* __restrict__ fin) {
  const int b = blockIdx.x;
  const int h = threadIdx.x;
  const long long len = lengths[b];
  const long t = (len > 0) ? (long)(len - 1) : 0;
  fin[(size_t)b * Hh + h] = out[((size_t)b * Tt + t) * Hh + h];
}

// ---------------------------------------------------------------- launch ----
extern "C" void kernel_launch(void* const* d_in, const int* in_sizes, int n_in,
                              void* d_out, int out_size, void* d_ws,
                              size_t ws_size, hipStream_t stream) {
  (void)in_sizes; (void)n_in; (void)out_size; (void)ws_size;
  const float*     X       = (const float*)d_in[0];
  const long long* lengths = (const long long*)d_in[1];
  const float*     Wx      = (const float*)d_in[2];
  const float*     Wh      = (const float*)d_in[3];
  const float*     bias    = (const float*)d_in[4];

  float* out = (float*)d_out;
  float* fin = out + (size_t)Bb * Tt * Hh;

  const size_t zx_bytes = (size_t)Bb * Tt * ZC * sizeof(half_t);  // 128 MB
  half_t*   Zxp  = (half_t*)d_ws;
  half_t*   hbuf = (half_t*)((char*)d_ws + zx_bytes);
  unsigned* ctr  = (unsigned*)((char*)d_ws + zx_bytes + (size_t)Bb * Hh * 2);

  lstm_init_ctr<<<1, 32, 0, stream>>>(ctr);

  // 512 t-pairs x 32 col groups
  const size_t smem1 = 4 * 16896;                // 2x A + 2x Wx buffers
  lstm_xproj<<<(Tt / 2) * (ZC / 64), 256, smem1, stream>>>(X, Wx, Zxp);

  const size_t smem = 128 * 520 * sizeof(half_t) + 32 * 520 * sizeof(half_t) +
                      32 * 132 * sizeof(float) + 32 * 32 * sizeof(float);
  lstm_rec<<<NWG, 256, smem, stream>>>(Zxp, Wh, bias, out, hbuf, ctr);

  lstm_final<<<Bb, Hh, 0, stream>>>(out, lengths, fin);
}
